// AdditiveAttention_1176821039630
// MI455X (gfx1250) — matrix operand order repaired and checked by reading, not compiled, not run on hardware
//
#include <hip/hip_runtime.h>
#include <stdint.h>

typedef __attribute__((ext_vector_type(16))) __bf16 v16bf;
typedef __attribute__((ext_vector_type(8)))  float  v8f;

#define Hdim 1024
#define Ldim 2048
#define Bdim 16
#define ROWS 64            // rows per workgroup in score kernel (4 m-tiles)
#define MT_TILES 4
#define KT_TILES 64        // combined K = 2048 -> 64 tiles of 32
#define KT_PER_PHASE 32    // K split in two LDS phases (key half / query half)
#define NWAVES 8
#define NT_PER_WAVE 8

union FragU { uint4 q[2]; v16bf v; };

__device__ __forceinline__ uint32_t pk_bf16(float a, float b) {
  uint32_t ua = __float_as_uint(a);
  uint32_t ub = __float_as_uint(b);
  ua = (ua + 0x7FFFu + ((ua >> 16) & 1u)) >> 16;   // RNE
  ub = (ub + 0x7FFFu + ((ub >> 16) & 1u)) >> 16;
  return (ua & 0xFFFFu) | (ub << 16);
}

// ---------------------------------------------------------------------------
// Pack [Wk | Wq] (fp32 [out,in]) into bf16 WMMA-B fragments.
// Fragment (nt, kt): 32 lanes x 8 dwords, lane L: n = nt*16 + L%16,
// dword v holds K rows { kt*32 + (L/16)*16 + 2v, +1 } packed lo|hi.
// ---------------------------------------------------------------------------
__global__ __launch_bounds__(256) void pack_w_kernel(
    const float* __restrict__ Wq, const float* __restrict__ Wk,
    uint32_t* __restrict__ wpack) {
  int t    = blockIdx.x * 256 + threadIdx.x;   // 4096 frags * 32 lanes
  int lane = t & 31;
  int frag = t >> 5;            // nt*64 + kt
  int kt   = frag & 63;
  int nt   = frag >> 6;
  int n     = nt * 16 + (lane & 15);
  int kbase = kt * 32 + (lane >> 4) * 16;
  uint32_t* dst = wpack + (size_t)frag * 256 + lane * 8;
#pragma unroll
  for (int v = 0; v < 8; ++v) {
    int k = kbase + 2 * v;                      // pair never straddles 1024
    const float* W = (k < Hdim) ? Wk : Wq;
    int kk = k & (Hdim - 1);
    float2 f = *(const float2*)(W + (size_t)n * Hdim + kk);
    dst[v] = pk_bf16(f.x, f.y);
  }
}

// ---------------------------------------------------------------------------
// Fused score kernel: h = tanh([key|query] @ [Wk|Wq]^T + bias); partial
// score[row] = sum_n h*w_score written per-wave (race-free) to sws[row*8+w].
// 64 rows per WG; K processed in two LDS phases (key half, then query half)
// so LDS stays at 128 KB while B-fragment L2 traffic is halved vs ROWS=32.
// ---------------------------------------------------------------------------
__global__ __launch_bounds__(256) void score_kernel(
    const float* __restrict__ key_t, const float* __restrict__ query,
    const uint32_t* __restrict__ wpack, const float* __restrict__ bias,
    const float* __restrict__ w_score, float* __restrict__ sws) {
  __shared__ uint32_t Abuf[ROWS * 512];   // 64 rows x 1024 bf16 = 128 KB

  const int tid     = threadIdx.x;
  const int rowBase = blockIdx.x * ROWS;

  const int w    = tid >> 5;
  const int lane = tid & 31;
  const int m0   = lane & 15;
  const int half = lane >> 4;

  v8f acc[MT_TILES][NT_PER_WAVE];
#pragma unroll
  for (int mt = 0; mt < MT_TILES; ++mt)
#pragma unroll
    for (int n = 0; n < NT_PER_WAVE; ++n)
      acc[mt][n] = (v8f){0.f, 0.f, 0.f, 0.f, 0.f, 0.f, 0.f, 0.f};

  for (int phase = 0; phase < 2; ++phase) {
    __syncthreads();   // all reads of previous phase's A done before restage
    const float* src = phase ? query : key_t;
    for (int i = tid; i < ROWS * 512; i += 256) {
      int r = i >> 9;             // row in block
      int p = i & 511;            // bf16 pair -> k = 2p, 2p+1 within half
      float2 f = *(const float2*)(src + (size_t)(rowBase + r) * Hdim + 2 * p);
      Abuf[r * 512 + p] = pk_bf16(f.x, f.y);
    }
    __syncthreads();

    for (int kt2 = 0; kt2 < KT_PER_PHASE; ++kt2) {
      int kt = phase * KT_PER_PHASE + kt2;   // global kt (B index)
      // A fragments: dword v -> k = half*16 + 2v
      FragU a[MT_TILES];
#pragma unroll
      for (int mt = 0; mt < MT_TILES; ++mt) {
        const uint32_t* pA = &Abuf[(mt * 16 + m0) * 512 + kt2 * 16 + half * 8];
        a[mt].q[0] = *(const uint4*)(pA);        // v0..v3
        a[mt].q[1] = *(const uint4*)(pA + 4);    // v4..v7
      }
#pragma unroll
      for (int ntl = 0; ntl < NT_PER_WAVE; ++ntl) {
        int NT = w * NT_PER_WAVE + ntl;
        const uint32_t* pB = wpack + ((size_t)(NT * 64 + kt) * 256 + lane * 8);
        FragU bfr;
        bfr.q[0] = *(const uint4*)(pB);
        bfr.q[1] = *(const uint4*)(pB + 4);
#pragma unroll
        for (int mt = 0; mt < MT_TILES; ++mt)
          acc[mt][ntl] = __builtin_amdgcn_wmma_f32_16x16x32_bf16(
              false, a[mt].v, false, bfr.v, (short)0, acc[mt][ntl], false, false);
      }
    }
  }

  // Epilogue: C layout -> lane holds column n = NT*16 + m0 for all 8 rows.
  float part[MT_TILES][8];
#pragma unroll
  for (int mt = 0; mt < MT_TILES; ++mt)
#pragma unroll
    for (int i = 0; i < 8; ++i) part[mt][i] = 0.f;

#pragma unroll
  for (int ntl = 0; ntl < NT_PER_WAVE; ++ntl) {
    int n = (w * NT_PER_WAVE + ntl) * 16 + m0;
    float bn  = bias[n];
    float wsn = w_score[n];
#pragma unroll
    for (int mt = 0; mt < MT_TILES; ++mt)
#pragma unroll
      for (int i = 0; i < 8; ++i)
        part[mt][i] += tanhf(acc[mt][ntl][i] + bn) * wsn;
  }

  // Reduce across each 16-lane half (rows m = i + 8*half within the m-tile).
#pragma unroll
  for (int mt = 0; mt < MT_TILES; ++mt)
#pragma unroll
    for (int i = 0; i < 8; ++i) {
      float p = part[mt][i];
      p += __shfl_xor(p, 1, 32);
      p += __shfl_xor(p, 2, 32);
      p += __shfl_xor(p, 4, 32);
      p += __shfl_xor(p, 8, 32);
      part[mt][i] = p;
    }
  if (m0 == 0) {
#pragma unroll
    for (int mt = 0; mt < MT_TILES; ++mt)
#pragma unroll
      for (int i = 0; i < 8; ++i) {
        int row = rowBase + mt * 16 + half * 8 + i;
        sws[(size_t)row * 8 + w] = part[mt][i];
      }
  }
}

// ---------------------------------------------------------------------------
// Softmax over L per batch (b_score is a constant shift -> dropped exactly).
// ---------------------------------------------------------------------------
__global__ __launch_bounds__(256) void softmax_kernel(
    const float* __restrict__ sws, float* __restrict__ attn) {
  __shared__ float sc[Ldim];
  __shared__ float red[256];
  int b = blockIdx.x, tid = threadIdx.x;

  float lmax = -3.4e38f;
  for (int l = tid; l < Ldim; l += 256) {
    const float* p = sws + (size_t)(b * Ldim + l) * 8;
    float s = ((p[0] + p[1]) + (p[2] + p[3])) + ((p[4] + p[5]) + (p[6] + p[7]));
    sc[l] = s;
    lmax = fmaxf(lmax, s);
  }
  red[tid] = lmax;
  __syncthreads();
  for (int o = 128; o > 0; o >>= 1) {
    if (tid < o) red[tid] = fmaxf(red[tid], red[tid + o]);
    __syncthreads();
  }
  float m = red[0];
  __syncthreads();

  float lsum = 0.f;
  for (int l = tid; l < Ldim; l += 256) {
    float e = expf(sc[l] - m);
    sc[l] = e;
    lsum += e;
  }
  red[tid] = lsum;
  __syncthreads();
  for (int o = 128; o > 0; o >>= 1) {
    if (tid < o) red[tid] += red[tid + o];
    __syncthreads();
  }
  float inv = 1.0f / red[0];
  for (int l = tid; l < Ldim; l += 256)
    attn[b * Ldim + l] = sc[l] * inv;
}

// ---------------------------------------------------------------------------
// context[b,h] = sum_l attn[b,l] * value[b,l,h]  (deterministic partials)
// ---------------------------------------------------------------------------
__global__ __launch_bounds__(256) void ctx_partial_kernel(
    const float* __restrict__ attn, const float* __restrict__ value,
    float* __restrict__ cpart) {
  int wg = blockIdx.x;             // 16 b * 8 lchunks * 4 hblocks
  int hb = wg & 3;
  int lc = (wg >> 2) & 7;
  int b  = wg >> 5;
  int h  = hb * 256 + threadIdx.x;
  int l0 = lc * 256;
  float acc = 0.f;
  for (int i = 0; i < 256; ++i) {
    int l = l0 + i;
    acc += attn[b * Ldim + l] * value[((size_t)(b * Ldim + l)) * Hdim + h];
  }
  cpart[(size_t)lc * (Bdim * Hdim) + b * Hdim + h] = acc;
}

__global__ __launch_bounds__(256) void ctx_reduce_kernel(
    const float* __restrict__ cpart, float* __restrict__ ctx) {
  int idx = blockIdx.x * 256 + threadIdx.x;   // b*Hdim + h, 16384 total
  float s = 0.f;
#pragma unroll
  for (int lc = 0; lc < 8; ++lc)
    s += cpart[(size_t)lc * (Bdim * Hdim) + idx];
  ctx[idx] = s;
}

// ---------------------------------------------------------------------------
extern "C" void kernel_launch(void* const* d_in, const int* in_sizes, int n_in,
                              void* d_out, int out_size, void* d_ws, size_t ws_size,
                              hipStream_t stream) {
  const float* query   = (const float*)d_in[0];
  const float* key_t   = (const float*)d_in[1];
  const float* value   = (const float*)d_in[2];
  const float* Wq      = (const float*)d_in[3];
  const float* Wk      = (const float*)d_in[4];
  const float* bias    = (const float*)d_in[5];
  const float* w_score = (const float*)d_in[6];
  (void)d_in[7]; (void)in_sizes; (void)n_in; (void)out_size; (void)ws_size;

  float* out  = (float*)d_out;
  float* ctx  = out;                 // [16,1,1024]
  float* attn = out + Bdim * Hdim;   // [16,2048]

  uint8_t*  ws    = (uint8_t*)d_ws;
  uint32_t* wpack = (uint32_t*)ws;                        // 4 MB packed bf16 W
  float*    sws   = (float*)(ws + (4u << 20));            // 1 MB score partials
  float*    cpart = (float*)(ws + (5u << 20));            // 512 KB ctx partials

  hipLaunchKernelGGL(pack_w_kernel, dim3(512), dim3(256), 0, stream, Wq, Wk, wpack);
  hipLaunchKernelGGL(score_kernel, dim3((Bdim * Ldim) / ROWS), dim3(256), 0, stream,
                     key_t, query, wpack, bias, w_score, sws);
  hipLaunchKernelGGL(softmax_kernel, dim3(Bdim), dim3(256), 0, stream, sws, attn);
  hipLaunchKernelGGL(ctx_partial_kernel, dim3(512), dim3(256), 0, stream, attn, value, cpart);
  hipLaunchKernelGGL(ctx_reduce_kernel, dim3(64), dim3(256), 0, stream, cpart, ctx);
}
